// TaskAlignedAssigner_3375844294784
// MI455X (gfx1250) — compile-verified, matrix operand order checked by reading, not run on hardware
//
#include <hip/hip_runtime.h>
#include <stdint.h>

#define TOPK 13

typedef _Float16 half16 __attribute__((ext_vector_type(16)));
typedef float    float8 __attribute__((ext_vector_type(8)));
typedef unsigned int u32x4 __attribute__((ext_vector_type(4)));
typedef int      i32x4  __attribute__((ext_vector_type(4)));
typedef int      i32x8  __attribute__((ext_vector_type(8)));

// CIoU (box1 = gt, box2 = pred), clipped at 0, matching the JAX reference.
__device__ __forceinline__ float ciou_clip(float gx1, float gy1, float gx2, float gy2,
                                           float px1, float py1, float px2, float py2) {
  const float eps = 1e-7f;
  float w1 = gx2 - gx1, h1 = gy2 - gy1 + eps;
  float w2 = px2 - px1, h2 = py2 - py1 + eps;
  float iw = fmaxf(fminf(gx2, px2) - fmaxf(gx1, px1), 0.f);
  float ih = fmaxf(fminf(gy2, py2) - fmaxf(gy1, py1), 0.f);
  float inter = iw * ih;
  float uni   = w1 * h1 + w2 * h2 - inter + eps;
  float iou   = inter / uni;
  float cw = fmaxf(gx2, px2) - fminf(gx1, px1);
  float ch = fmaxf(gy2, py2) - fminf(gy1, py1);
  float c2 = cw * cw + ch * ch + eps;
  float dx = px1 + px2 - gx1 - gx2, dy = py1 + py2 - gy1 - gy2;
  float rho2 = (dx * dx + dy * dy) * 0.25f;
  float dv = atanf(w2 / h2) - atanf(w1 / h1);
  float v  = 0.4052847345693511f * dv * dv;     // 4/pi^2
  float alpha = v / (v - iou + 1.0f + eps);
  return fmaxf(iou - (rho2 / c2 + v * alpha), 0.f);
}

// ---------------------------------------------------------------------------
// K1: per (b,m) row — exact top-13 of align_metric*mask_in_gts over A anchors,
//     scatter 1.0h into dense mask. anc_points staged into LDS via TDM.
// ---------------------------------------------------------------------------
__global__ __launch_bounds__(256)
void k1_topk(const float* __restrict__ pd_scores,
             const float* __restrict__ pd_bboxes,
             const float* __restrict__ anc,
             const int*   __restrict__ gt_labels,
             const float* __restrict__ gt_bboxes,
             const float* __restrict__ mask_gt,
             unsigned short* __restrict__ maskh,
             int A, int M, int NC) {
  extern __shared__ char smem[];
  float* s_anc = (float*)smem;                 // 2*A floats
  float* sv    = s_anc + 2 * A;                // 256*TOPK
  int*   sa    = (int*)(sv + 256 * TOPK);      // 256*TOPK
  float* rv    = (float*)(sa + 256 * TOPK);    // 256
  int*   ra    = (int*)(rv + 256);             // 256
  int*   rs    = ra + 256;                     // 256
  int*   sel   = rs + 256;                     // TOPK

  const int m = blockIdx.x, b = blockIdx.y;
  const int tid = threadIdx.x;
  if (mask_gt[b * M + m] <= 0.f) return;       // uniform: invalid GT row stays 0

  // ---- stage anchor table into LDS (Tensor Data Mover on gfx1250) ----
#if defined(__gfx1250__) && __has_builtin(__builtin_amdgcn_tensor_load_to_lds) && \
    __has_builtin(__builtin_amdgcn_s_wait_tensorcnt)
  if (tid < 32) {  // one wave issues the DMA; TDM ignores EXEC
    uint64_t ga = (uint64_t)(uintptr_t)anc;
    unsigned lds_off = (unsigned)(uintptr_t)s_anc;   // low 32 bits = LDS byte addr
    unsigned ne = (unsigned)(2 * A);                 // elements (4B each)
    // D# group0: count=1 | lds_addr | global_addr[56:0] | type=2
    u32x4 g0 = { 1u, lds_off, (unsigned)(ga & 0xffffffffu),
                 (unsigned)((ga >> 32) & 0x1ffffffu) | 0x80000000u };
    // D# group1: data_size=4B, tensor_dim0=ne, tensor_dim1=1, tile_dim0=ne,
    //            tile_dim1=1, tensor_dim0_stride=ne
    i32x8 g1 = { (int)(2u << 16),
                 (int)((ne & 0xffffu) << 16),
                 (int)(((ne >> 16) & 0xffffu) | (1u << 16)),
                 (int)((ne & 0xffffu) << 16),
                 1, (int)ne, 0, 0 };
    i32x4 z4 = {0, 0, 0, 0};
#if defined(__clang_major__) && __clang_major__ >= 23
    i32x8 z8 = {0, 0, 0, 0, 0, 0, 0, 0};
    __builtin_amdgcn_tensor_load_to_lds(g0, g1, z4, z4, z8, 0);
#else
    __builtin_amdgcn_tensor_load_to_lds(g0, g1, z4, z4, 0);
#endif
    __builtin_amdgcn_s_wait_tensorcnt(0);
  }
#else
  for (int i = tid; i < 2 * A; i += 256) s_anc[i] = anc[i];
#endif
  __syncthreads();

  const float4 g  = reinterpret_cast<const float4*>(gt_bboxes)[b * M + m];
  const int    lbl = gt_labels[b * M + m];
  const float4* pb4 = reinterpret_cast<const float4*>(pd_bboxes) + (size_t)b * A;
  const float*  sc  = pd_scores + (size_t)b * A * NC + lbl;

  // ---- per-thread top-13 over strided anchors ----
  float lv[TOPK]; int la[TOPK];
#pragma unroll
  for (int i = 0; i < TOPK; ++i) { lv[i] = -1.f; la[i] = 0x7fffffff; }

  for (int a = tid; a < A; a += 256) {
    float ax = s_anc[2 * a], ay = s_anc[2 * a + 1];
    float dmin = fminf(fminf(ax - g.x, ay - g.y), fminf(g.z - ax, g.w - ay));
    float4 p = pb4[a];
    float o  = ciou_clip(g.x, g.y, g.z, g.w, p.x, p.y, p.z, p.w);
    float s  = sc[(size_t)a * NC];
    float o2 = o * o;
    float met = (dmin > 1e-9f) ? (s * o2 * o2 * o2) : 0.f;  // alpha=1, beta=6
    if (met > lv[TOPK - 1] || (met == lv[TOPK - 1] && a < la[TOPK - 1])) {
      lv[TOPK - 1] = met; la[TOPK - 1] = a;
#pragma unroll
      for (int j = TOPK - 1; j > 0; --j) {
        bool up = (lv[j] > lv[j - 1]) || (lv[j] == lv[j - 1] && la[j] < la[j - 1]);
        if (!up) break;
        float tv = lv[j]; lv[j] = lv[j - 1]; lv[j - 1] = tv;
        int   ti = la[j]; la[j] = la[j - 1]; la[j - 1] = ti;
      }
    }
  }
#pragma unroll
  for (int i = 0; i < TOPK; ++i) { sv[tid * TOPK + i] = lv[i]; sa[tid * TOPK + i] = la[i]; }
  __syncthreads();

  // ---- 13 rounds of block-wide argmax-with-removal (stable ties: low index) ----
  for (int r = 0; r < TOPK; ++r) {
    float bv = -2.f; int ba = 0x7fffffff; int bslot = -1;
#pragma unroll
    for (int i = 0; i < TOPK; ++i) {
      int j = tid * TOPK + i;
      float v = sv[j]; int aa = sa[j];
      if (v > bv || (v == bv && aa < ba)) { bv = v; ba = aa; bslot = j; }
    }
    rv[tid] = bv; ra[tid] = ba; rs[tid] = bslot;
    __syncthreads();
    for (int s = 128; s > 0; s >>= 1) {
      if (tid < s) {
        float v2 = rv[tid + s]; int a2 = ra[tid + s];
        if (v2 > rv[tid] || (v2 == rv[tid] && a2 < ra[tid])) {
          rv[tid] = v2; ra[tid] = a2; rs[tid] = rs[tid + s];
        }
      }
      __syncthreads();
    }
    if (tid == 0) { sel[r] = ra[0]; sv[rs[0]] = -3.f; sa[rs[0]] = 0x7fffffff; }
    __syncthreads();
  }

  // ---- scatter: mask_pos0 = topk ∧ in_gts ∧ mask_gt ----
  if (tid < TOPK) {
    int a = sel[tid];
    if (a >= 0 && a < A) {
      float ax = s_anc[2 * a], ay = s_anc[2 * a + 1];
      float dmin = fminf(fminf(ax - g.x, ay - g.y), fminf(g.z - ax, g.w - ay));
      if (dmin > 1e-9f)
        maskh[(size_t)(b * M + m) * A + a] = 0x3C00;  // f16 1.0
    }
  }
}

// ---------------------------------------------------------------------------
// K2: fg count over M via V_WMMA_F32_16X16X32_F16 (A-operand = ones-row), then
//     per-anchor conflict resolution. 8 waves/block, one 16-anchor tile/wave.
// ---------------------------------------------------------------------------
__global__ __launch_bounds__(256)
void k2_resolve(const float* __restrict__ pd_bboxes,
                const float* __restrict__ gt_bboxes,
                const unsigned short* __restrict__ maskh,
                float* __restrict__ fg_f, int* __restrict__ tgt_i,
                int A, int M) {
  const int lane = threadIdx.x & 31;
  const int wave = threadIdx.x >> 5;
  const int tile = blockIdx.x * 8 + wave;           // 16 anchors per tile
  const int ntiles = (A + 15) / 16;
  if (tile >= ntiles) return;                       // wave-uniform
  const int a0 = tile * 16;
  const int b  = blockIdx.y;
  const _Float16* mh = reinterpret_cast<const _Float16*>(maskh);
  const size_t base = (size_t)b * M * A;

  const int col  = lane & 15;                       // N column (row across lanes)
  const int kgrp = (lane >> 4) << 4;                // lanes 16-31: upper K half
  const int ac   = (a0 + col < A) ? (a0 + col) : (A - 1);
  const int Mfull = M & ~31;                        // full 32-row K groups

  float cnt = 0.f;
#if defined(__gfx1250__)
  {
    float8 c;
#pragma unroll
    for (int i = 0; i < 8; ++i) c[i] = 0.f;
    half16 av;
    _Float16 one = (col == 0) ? (_Float16)1.0f : (_Float16)0.0f; // row M=0 = ones
#pragma unroll
    for (int t = 0; t < 16; ++t) av[t] = one;

    // running pointer: no per-element mul, no per-element guards
    const _Float16* q = mh + base + (size_t)kgrp * A + ac;
    const size_t strideA = (size_t)A;
    for (int ks = 0; ks < Mfull; ks += 32) {
      half16 bv;
#pragma unroll
      for (int t = 0; t < 16; ++t) { bv[t] = *q; q += strideA; }
      q += 16 * strideA;                            // skip other lane-group half
      c = __builtin_amdgcn_wmma_f32_16x16x32_f16(false, av, false, bv,
                                                 (short)0, c, false, false);
    }
    cnt = c[0];  // D row M=0: VGPR0, lanes 0..15 = N
  }
#else
  if (lane < 16 && a0 + lane < A)
    for (int mrow = 0; mrow < Mfull; ++mrow)
      cnt += (float)mh[base + (size_t)mrow * A + (a0 + lane)];
#endif

  if (lane < 16 && a0 + lane < A) {
    const int a = a0 + lane;
    for (int mrow = Mfull; mrow < M; ++mrow)        // remainder rows (M%32)
      cnt += (float)mh[base + (size_t)mrow * A + a];
    const int n = (int)(cnt + 0.5f);
    int tgt = 0; float fg = 0.f;
    if (n == 1) {
      for (int mrow = 0; mrow < M; ++mrow)
        if (maskh[base + (size_t)mrow * A + a] != 0) { tgt = mrow; break; }
      fg = 1.f;
    } else if (n > 1) {  // mask_multi -> one-hot of argmax_m overlaps
      float4 p = reinterpret_cast<const float4*>(pd_bboxes)[(size_t)b * A + a];
      float best = -1.f; int bm = 0;
      for (int mrow = 0; mrow < M; ++mrow) {
        float4 gg = reinterpret_cast<const float4*>(gt_bboxes)[b * M + mrow];
        float o = ciou_clip(gg.x, gg.y, gg.z, gg.w, p.x, p.y, p.z, p.w);
        if (o > best) { best = o; bm = mrow; }
      }
      tgt = bm; fg = 1.f;
    }
    fg_f[(size_t)b * A + a] = fg;
    tgt_i[(size_t)b * A + a] = tgt;
  }
}

// ---------------------------------------------------------------------------
// K3: per fg anchor — align_metric / overlap at (b, tgt, a); bitwise atomicMax
//     (non-negative floats) builds pos_align / pos_overlaps per (b, m).
// ---------------------------------------------------------------------------
__global__ __launch_bounds__(256)
void k3_stats(const float* __restrict__ pd_scores,
              const float* __restrict__ pd_bboxes,
              const int*   __restrict__ gt_labels,
              const float* __restrict__ gt_bboxes,
              const float* __restrict__ fg_f, const int* __restrict__ tgt_i,
              float* __restrict__ align_pa,
              unsigned int* __restrict__ pos_al, unsigned int* __restrict__ pos_ov,
              int A, int M, int NC, int bs) {
  int id = blockIdx.x * blockDim.x + threadIdx.x;
  if (id >= bs * A) return;
  if (fg_f[id] <= 0.f) return;
  int b = id / A;
  int tgt = tgt_i[id];
  float4 g = reinterpret_cast<const float4*>(gt_bboxes)[b * M + tgt];
  float4 p = reinterpret_cast<const float4*>(pd_bboxes)[id];
  float o = ciou_clip(g.x, g.y, g.z, g.w, p.x, p.y, p.z, p.w);
  int lbl = gt_labels[b * M + tgt];
  float s = pd_scores[(size_t)id * NC + lbl];
  float o2 = o * o;
  float al = s * o2 * o2 * o2;
  align_pa[id] = al;
  atomicMax(&pos_al[b * M + tgt], __float_as_uint(al));
  atomicMax(&pos_ov[b * M + tgt], __float_as_uint(o));
}

// ---------------------------------------------------------------------------
// K4: fused output writer — labels, bboxes, scores(one-hot * norm), fg, gt_idx.
// ---------------------------------------------------------------------------
__global__ __launch_bounds__(256)
void k4_out(const int*   __restrict__ gt_labels,
            const float* __restrict__ gt_bboxes,
            const float* __restrict__ fg_f, const int* __restrict__ tgt_i,
            const float* __restrict__ align_pa,
            const unsigned int* __restrict__ pos_al, const unsigned int* __restrict__ pos_ov,
            float* __restrict__ out, int A, int M, int NC, int bs) {
  int id = blockIdx.x * blockDim.x + threadIdx.x;
  if (id >= bs * A) return;
  int b = id / A;
  int tgt = tgt_i[id];
  float fg = fg_f[id];
  int tl = gt_labels[b * M + tgt]; tl = tl < 0 ? 0 : tl;

  size_t bsA = (size_t)bs * A;
  out[id] = (float)tl;                                              // target_labels
  reinterpret_cast<float4*>(out + bsA)[id] =
      reinterpret_cast<const float4*>(gt_bboxes)[b * M + tgt];      // target_bboxes

  float norm = 0.f;
  if (fg > 0.f) {
    float pa = __uint_as_float(pos_al[b * M + tgt]);
    float po = __uint_as_float(pos_ov[b * M + tgt]);
    norm = align_pa[id] * po / (pa + 1e-9f);
  }
  float* srow = out + bsA * 5 + (size_t)id * NC;                    // target_scores
  float4 z = make_float4(0.f, 0.f, 0.f, 0.f);
  for (int c = 0; c < NC; c += 4) reinterpret_cast<float4*>(srow + c)[0] = z;
  if (fg > 0.f) srow[tl] = norm;

  size_t o_fg = bsA * (5 + NC);
  out[o_fg + id]       = fg;                                        // fg_mask
  out[o_fg + bsA + id] = (float)tgt;                                // target_gt_idx
}

// ---------------------------------------------------------------------------
extern "C" void kernel_launch(void* const* d_in, const int* in_sizes, int n_in,
                              void* d_out, int out_size, void* d_ws, size_t ws_size,
                              hipStream_t stream) {
  const float* pd_scores = (const float*)d_in[0];
  const float* pd_bboxes = (const float*)d_in[1];
  const float* anc       = (const float*)d_in[2];
  const int*   gt_labels = (const int*)d_in[3];
  const float* gt_bboxes = (const float*)d_in[4];
  const float* mask_gt   = (const float*)d_in[5];

  const int A  = in_sizes[2] / 2;
  const int bs = in_sizes[1] / (4 * A);
  const int M  = in_sizes[3] / bs;
  const int NC = in_sizes[0] / (bs * A);

  char* ws = (char*)d_ws;
  size_t off = 0;
  auto alloc = [&](size_t bytes) -> void* {
    void* p = ws + off; off += (bytes + 255) & ~(size_t)255; return p;
  };
  unsigned short* maskh   = (unsigned short*)alloc((size_t)bs * M * A * 2);
  float*          fg_f    = (float*)alloc((size_t)bs * A * 4);
  int*            tgt_i   = (int*)alloc((size_t)bs * A * 4);
  float*          align_pa= (float*)alloc((size_t)bs * A * 4);
  unsigned int*   pos_al  = (unsigned int*)alloc((size_t)bs * M * 4);
  unsigned int*   pos_ov  = (unsigned int*)alloc((size_t)bs * M * 4);

  hipMemsetAsync(maskh, 0, (size_t)bs * M * A * 2, stream);
  hipMemsetAsync(pos_al, 0, (size_t)bs * M * 4, stream);
  hipMemsetAsync(pos_ov, 0, (size_t)bs * M * 4, stream);

  size_t smem = (size_t)(2 * A) * sizeof(float)
              + 256 * TOPK * (sizeof(float) + sizeof(int))
              + 256 * (sizeof(float) + 2 * sizeof(int))
              + TOPK * sizeof(int) + 64;

  k1_topk<<<dim3(M, bs), 256, smem, stream>>>(pd_scores, pd_bboxes, anc,
                                              gt_labels, gt_bboxes, mask_gt,
                                              maskh, A, M, NC);
  int ntiles = (A + 15) / 16;
  k2_resolve<<<dim3((ntiles + 7) / 8, bs), 256, 0, stream>>>(pd_bboxes, gt_bboxes,
                                                             maskh, fg_f, tgt_i, A, M);
  int n = bs * A;
  k3_stats<<<(n + 255) / 256, 256, 0, stream>>>(pd_scores, pd_bboxes, gt_labels,
                                                gt_bboxes, fg_f, tgt_i, align_pa,
                                                pos_al, pos_ov, A, M, NC, bs);
  k4_out<<<(n + 255) / 256, 256, 0, stream>>>(gt_labels, gt_bboxes, fg_f, tgt_i,
                                              align_pa, pos_al, pos_ov,
                                              (float*)d_out, A, M, NC, bs);
}
